// TNetwork_17454747091444
// MI455X (gfx1250) — matrix operand order repaired and checked by reading, not compiled, not run on hardware
//
#include <hip/hip_runtime.h>

// ---------------------------------------------------------------------------
// CDNA5 (gfx1250) GCN pipeline: WMMA bf16 GEMMs + L2-resident edge scatter.
// ---------------------------------------------------------------------------

typedef __attribute__((ext_vector_type(16))) __bf16 bfx16;
typedef __attribute__((ext_vector_type(8)))  float  fx8;

#define FEAT 128   // IN_C == HID == 128

// ------------------------------ small utility kernels ----------------------

__global__ void fill_f32(float* __restrict__ p, float v, int n) {
    int t = blockIdx.x * blockDim.x + threadIdx.x;
    if (t < n) p[t] = v;
}

__global__ void deg_accum(const int* __restrict__ dst, float* __restrict__ deg, int E) {
    int t = blockIdx.x * blockDim.x + threadIdx.x;
    if (t < E) atomicAdd(deg + dst[t], 1.0f);
}

__global__ void rsqrt_ip(float* __restrict__ p, int n) {
    int t = blockIdx.x * blockDim.x + threadIdx.x;
    if (t < n) p[t] = rsqrtf(p[t]);
}

// Pack W[K][N] (f32 row-major) into the exact per-lane bf16 B-matrix layout
// for v_wmma_f32_16x16x32_bf16:
//   element e of lane (n + 16*hi) holds B[kb*32 + hi*8 + (e&7) + (e>>3)*16][ntile*16 + n]
// Stored as [ntile][kb][lane][e] so each GEMM lane does one contiguous 32B load.
__global__ void pack_w_bf16(const float* __restrict__ W, __bf16* __restrict__ Wp,
                            int K, int Ncols) {
    int t = blockIdx.x * blockDim.x + threadIdx.x;
    int total = K * Ncols;
    if (t >= total) return;
    int e      = t & 15;
    int lane   = (t >> 4) & 31;
    int blk    = t >> 9;               // ntile * (K/32) + kb
    int kbcnt  = K >> 5;
    int kb     = blk % kbcnt;
    int ntile  = blk / kbcnt;
    int n      = (ntile << 4) + (lane & 15);
    int k      = (kb << 5) + ((lane >> 4) << 3) + (e & 7) + ((e >> 3) << 4);
    Wp[t] = (__bf16)W[(size_t)k * Ncols + n];
}

// ------------------------------ WMMA GEMM ----------------------------------
// C[M][Ncols] = act(A[M][K]) @ Wpacked + bias   (act = optional ReLU on A)
// One wave computes one 16x16 tile; 8 waves/block cover 8 n-tiles.
// KB = K/32 and RELU are compile-time -> fully unrolled WMMA stream, no
// per-iteration branch; WMMA (tracked as TRANS) co-executes with cvt VALU.
// A is f32 row-major, converted to bf16 in-register per the 16-bit A layout:
//   lane<16 : M = lane,    e0..7 -> K = kb*32 + 0..7,  e8..15 -> K = kb*32+16..23
//   lane>=16: M = lane-16, e0..7 -> K = kb*32 + 8..15, e8..15 -> K = kb*32+24..31
template <int KB, int RELU>
__global__ __launch_bounds__(256)
void wmma_gemm(const float* __restrict__ A, const __bf16* __restrict__ Bp,
               const float* __restrict__ bias, float* __restrict__ C,
               int Ncols) {
    const int lane   = threadIdx.x & 31;
    const int wave   = threadIdx.x >> 5;
    const int ntiles = Ncols >> 4;
    const int ntile  = blockIdx.y * 8 + wave;
    if (ntile >= ntiles) return;               // wave-uniform: EXEC stays all-1s
    const int hi     = lane >> 4;
    const int K      = KB * 32;
    const int mrow   = (blockIdx.x << 4) + (lane & 15);

    fx8 acc = {0.f, 0.f, 0.f, 0.f, 0.f, 0.f, 0.f, 0.f};
    const float*  arow  = A + (size_t)mrow * K + hi * 8;
    const __bf16* bbase = Bp + (size_t)ntile * KB * 512 + (size_t)lane * 16;

#pragma unroll
    for (int kb = 0; kb < KB; ++kb) {
        const float* ap = arow + kb * 32;
        float4 p0 = *(const float4*)(ap);
        float4 p1 = *(const float4*)(ap + 4);
        float4 p2 = *(const float4*)(ap + 16);
        float4 p3 = *(const float4*)(ap + 20);
        if (RELU) {
            p0.x = fmaxf(p0.x, 0.f); p0.y = fmaxf(p0.y, 0.f); p0.z = fmaxf(p0.z, 0.f); p0.w = fmaxf(p0.w, 0.f);
            p1.x = fmaxf(p1.x, 0.f); p1.y = fmaxf(p1.y, 0.f); p1.z = fmaxf(p1.z, 0.f); p1.w = fmaxf(p1.w, 0.f);
            p2.x = fmaxf(p2.x, 0.f); p2.y = fmaxf(p2.y, 0.f); p2.z = fmaxf(p2.z, 0.f); p2.w = fmaxf(p2.w, 0.f);
            p3.x = fmaxf(p3.x, 0.f); p3.y = fmaxf(p3.y, 0.f); p3.z = fmaxf(p3.z, 0.f); p3.w = fmaxf(p3.w, 0.f);
        }
        bfx16 a;
        a[0]  = (__bf16)p0.x; a[1]  = (__bf16)p0.y; a[2]  = (__bf16)p0.z; a[3]  = (__bf16)p0.w;
        a[4]  = (__bf16)p1.x; a[5]  = (__bf16)p1.y; a[6]  = (__bf16)p1.z; a[7]  = (__bf16)p1.w;
        a[8]  = (__bf16)p2.x; a[9]  = (__bf16)p2.y; a[10] = (__bf16)p2.z; a[11] = (__bf16)p2.w;
        a[12] = (__bf16)p3.x; a[13] = (__bf16)p3.y; a[14] = (__bf16)p3.z; a[15] = (__bf16)p3.w;

        bfx16 b = *(const bfx16*)(bbase + (size_t)kb * 512);

        acc = __builtin_amdgcn_wmma_f32_16x16x32_bf16(
                  false, a, false, b, (short)0, acc, false, false);
    }

    // C/D layout: lane<16 -> M = r, lane>=16 -> M = 8+r; N = lane&15.
    const int col = (ntile << 4) + (lane & 15);
    const float bv = bias ? bias[col] : 0.f;
    const int r0 = (blockIdx.x << 4) + (hi ? 8 : 0);
#pragma unroll
    for (int r = 0; r < 8; ++r)
        C[(size_t)(r0 + r) * Ncols + col] = acc[r] + bv;
}

// ------------------------------ GCN propagate ------------------------------

// agg[i][:] = h[i][:] * dinv[i]^2 + bias   (self-loop term, full overwrite)
__global__ void selfloop_bias(const float* __restrict__ h, const float* __restrict__ dinv,
                              const float* __restrict__ b, float* __restrict__ agg, int N) {
    int t = blockIdx.x * blockDim.x + threadIdx.x;   // N*32 threads, float4 each
    if (t >= N * 32) return;
    int node = t >> 5, q = t & 31;
    float di = dinv[node];
    float w  = di * di;
    float4 v  = ((const float4*)(h + (size_t)node * FEAT))[q];
    float4 bb = ((const float4*)b)[q];
    float4 r;
    r.x = v.x * w + bb.x; r.y = v.y * w + bb.y;
    r.z = v.z * w + bb.z; r.w = v.w * w + bb.w;
    ((float4*)(agg + (size_t)node * FEAT))[q] = r;
}

// One wave per edge: agg[dst] += h[src] * dinv[src]*dinv[dst]
// h and agg are ~51 MB each -> L2-resident; atomics resolve in L2.
__global__ __launch_bounds__(256)
void gcn_scatter(const int* __restrict__ src, const int* __restrict__ dst,
                 const float* __restrict__ dinv, const float* __restrict__ h,
                 float* __restrict__ agg, int E) {
    int gid  = blockIdx.x * blockDim.x + threadIdx.x;
    int edge = gid >> 5;
    int lane = gid & 31;
    if (edge >= E) return;
    int s = src[edge], d = dst[edge];
    float nrm = dinv[s] * dinv[d];
    float4 v = ((const float4*)(h + (size_t)s * FEAT))[lane];
    float* ad = agg + (size_t)d * FEAT + lane * 4;
    atomicAdd(ad + 0, v.x * nrm);
    atomicAdd(ad + 1, v.y * nrm);
    atomicAdd(ad + 2, v.z * nrm);
    atomicAdd(ad + 3, v.w * nrm);
}

// ------------------------------ pooling ------------------------------------

__global__ __launch_bounds__(256)
void pool_accum(const float* __restrict__ h, const int* __restrict__ batch,
                float* __restrict__ fsum, float* __restrict__ cnt, int N) {
    int gid  = blockIdx.x * blockDim.x + threadIdx.x;
    int node = gid >> 5;
    int lane = gid & 31;
    if (node >= N) return;
    int g = batch[node];
    float4 v = ((const float4*)(h + (size_t)node * FEAT))[lane];
    float* fs = fsum + (size_t)g * FEAT + lane * 4;
    atomicAdd(fs + 0, v.x);
    atomicAdd(fs + 1, v.y);
    atomicAdd(fs + 2, v.z);
    atomicAdd(fs + 3, v.w);
    if (lane == 0) atomicAdd(cnt + g, 1.0f);
}

__global__ void pool_div(const float* __restrict__ fsum, const float* __restrict__ cnt,
                         float* __restrict__ out, float* __restrict__ fbuf, int total) {
    int t = blockIdx.x * blockDim.x + threadIdx.x;
    if (t >= total) return;
    int g = t >> 7;   // FEAT == 128
    float c = fmaxf(cnt[g], 1.0f);
    float v = fsum[t] / c;
    out[t]  = v;      // output 0: pooled features f
    fbuf[t] = v;      // MLP input
}

// ------------------------------ host side ----------------------------------

static inline size_t align256(size_t x) { return (x + 255) & ~(size_t)255; }
static inline int cdiv(int a, int b) { return (a + b - 1) / b; }

// Launch dispatcher over the templated GEMM (K in {128,256,512}).
static void launch_gemm(const float* A, const __bf16* Wp, const float* bias, float* C,
                        int M, int K, int Ncols, int reluA, hipStream_t stream) {
    dim3 g(M / 16, cdiv(Ncols / 16, 8));
    dim3 b(256);
    switch ((K >> 5) * 2 + (reluA ? 1 : 0)) {
        case 4*2+0:  wmma_gemm<4,  0><<<g, b, 0, stream>>>(A, Wp, bias, C, Ncols); break;
        case 4*2+1:  wmma_gemm<4,  1><<<g, b, 0, stream>>>(A, Wp, bias, C, Ncols); break;
        case 8*2+0:  wmma_gemm<8,  0><<<g, b, 0, stream>>>(A, Wp, bias, C, Ncols); break;
        case 8*2+1:  wmma_gemm<8,  1><<<g, b, 0, stream>>>(A, Wp, bias, C, Ncols); break;
        case 16*2+0: wmma_gemm<16, 0><<<g, b, 0, stream>>>(A, Wp, bias, C, Ncols); break;
        case 16*2+1: wmma_gemm<16, 1><<<g, b, 0, stream>>>(A, Wp, bias, C, Ncols); break;
    }
}

extern "C" void kernel_launch(void* const* d_in, const int* in_sizes, int n_in,
                              void* d_out, int out_size, void* d_ws, size_t ws_size,
                              hipStream_t stream) {
    (void)n_in; (void)ws_size;

    const float* x     = (const float*)d_in[0];
    const int*   ei    = (const int*)d_in[1];
    const int*   batch = (const int*)d_in[2];
    const float* W1  = (const float*)d_in[3];  const float* b1  = (const float*)d_in[4];
    const float* W2  = (const float*)d_in[5];  const float* b2  = (const float*)d_in[6];
    const float* W3  = (const float*)d_in[7];  const float* b3  = (const float*)d_in[8];
    const float* Wf1 = (const float*)d_in[9];  const float* bf1 = (const float*)d_in[10];
    const float* Wf2 = (const float*)d_in[11]; const float* bf2 = (const float*)d_in[12];
    const float* Wf3 = (const float*)d_in[13]; const float* bf3 = (const float*)d_in[14];
    const float* Wf4 = (const float*)d_in[15]; const float* bf4 = (const float*)d_in[16];

    const int N = in_sizes[0] / FEAT;     // 100000 (multiple of 16)
    const int E = in_sizes[1] / 2;        // 1600000
    const int G = out_size / (FEAT + 64); // 512
    const int* src = ei;
    const int* dst = ei + E;
    float* out = (float*)d_out;

    // ---- workspace bump allocator ----
    char* ws = (char*)d_ws; size_t off = 0;
    auto alloc = [&](size_t bytes) -> void* {
        void* p = ws + off; off = align256(off + bytes); return p;
    };
    float* h_lin = (float*)alloc((size_t)N * FEAT * 4);
    float* agg_a = (float*)alloc((size_t)N * FEAT * 4);
    float* agg_b = (float*)alloc((size_t)N * FEAT * 4);
    float* dinv  = (float*)alloc((size_t)N * 4);           // degree -> rsqrt in place
    float* fsum  = (float*)alloc((size_t)G * FEAT * 4);
    float* cnt   = (float*)alloc((size_t)G * 4);
    float* fbuf  = (float*)alloc((size_t)G * FEAT * 4);
    float* z1    = (float*)alloc((size_t)G * 512 * 4);
    float* z2    = (float*)alloc((size_t)G * 256 * 4);
    float* z3    = (float*)alloc((size_t)G * 128 * 4);
    __bf16* W1p  = (__bf16*)alloc((size_t)128 * 128 * 2);
    __bf16* W2p  = (__bf16*)alloc((size_t)128 * 128 * 2);
    __bf16* W3p  = (__bf16*)alloc((size_t)128 * 128 * 2);
    __bf16* Wf1p = (__bf16*)alloc((size_t)128 * 512 * 2);
    __bf16* Wf2p = (__bf16*)alloc((size_t)512 * 256 * 2);
    __bf16* Wf3p = (__bf16*)alloc((size_t)256 * 128 * 2);
    __bf16* Wf4p = (__bf16*)alloc((size_t)128 * 64 * 2);

    const int B = 256;

    // ---- degree normalization: deg = 1 + scatter(1); dinv = rsqrt(deg) ----
    fill_f32<<<cdiv(N, B), B, 0, stream>>>(dinv, 1.0f, N);
    deg_accum<<<cdiv(E, B), B, 0, stream>>>(dst, dinv, E);
    rsqrt_ip<<<cdiv(N, B), B, 0, stream>>>(dinv, N);

    // ---- pack all weights into WMMA B layout (bf16) ----
    pack_w_bf16<<<cdiv(128 * 128, B), B, 0, stream>>>(W1,  W1p,  128, 128);
    pack_w_bf16<<<cdiv(128 * 128, B), B, 0, stream>>>(W2,  W2p,  128, 128);
    pack_w_bf16<<<cdiv(128 * 128, B), B, 0, stream>>>(W3,  W3p,  128, 128);
    pack_w_bf16<<<cdiv(128 * 512, B), B, 0, stream>>>(Wf1, Wf1p, 128, 512);
    pack_w_bf16<<<cdiv(512 * 256, B), B, 0, stream>>>(Wf2, Wf2p, 512, 256);
    pack_w_bf16<<<cdiv(256 * 128, B), B, 0, stream>>>(Wf3, Wf3p, 256, 128);
    pack_w_bf16<<<cdiv(128 * 64,  B), B, 0, stream>>>(Wf4, Wf4p, 128, 64);

    // ---- GCN layer helper (ReLU of previous layer folded into GEMM A-read) --
    auto gcn_layer = [&](const float* in, const __bf16* Wp, const float* bias,
                         float* agg, int reluA) {
        launch_gemm(in, Wp, nullptr, h_lin, N, FEAT, FEAT, reluA, stream);
        selfloop_bias<<<cdiv(N * 32, B), B, 0, stream>>>(h_lin, dinv, bias, agg, N);
        gcn_scatter<<<cdiv(E * 32, B), B, 0, stream>>>(src, dst, dinv, h_lin, agg, E);
    };
    gcn_layer(x,     W1p, b1, agg_a, 0);   // layer 1 (input x, no ReLU)
    gcn_layer(agg_a, W2p, b2, agg_b, 1);   // layer 2 (ReLU(agg_a) @ W2)
    gcn_layer(agg_b, W3p, b3, agg_a, 1);   // layer 3 (ReLU(agg_b) @ W3) -> agg_a

    // ---- global mean pool -> out[0 : G*128] and fbuf ----
    fill_f32<<<cdiv(G * FEAT, B), B, 0, stream>>>(fsum, 0.0f, G * FEAT);
    fill_f32<<<cdiv(G, B), B, 0, stream>>>(cnt, 0.0f, G);
    pool_accum<<<cdiv(N * 32, B), B, 0, stream>>>(agg_a, batch, fsum, cnt, N);
    pool_div<<<cdiv(G * FEAT, B), B, 0, stream>>>(fsum, cnt, out, fbuf, G * FEAT);

    // ---- MLP head (ReLU folded into next GEMM's A-read) ----
    launch_gemm(fbuf, Wf1p, bf1, z1, G, 128, 512, 0, stream);
    launch_gemm(z1,   Wf2p, bf2, z2, G, 512, 256, 1, stream);
    launch_gemm(z2,   Wf3p, bf3, z3, G, 256, 128, 1, stream);
    launch_gemm(z3,   Wf4p, bf4, out + (size_t)G * FEAT, G, 128, 64, 1, stream);
}
